// HybridSamplerQNN_65481071406467
// MI455X (gfx1250) — 8-bit matrix operand order repaired and checked by reading, compiled, not run on hardware
//
#include <hip/hip_runtime.h>
#include <hip/hip_bf16.h>
#include <math.h>

#define BATCH 256
#define SEQ   2048
#define INDIM 4
#define HID   256
#define GATES (4*HID)
#define WHH_BYTES (GATES*HID)

typedef __attribute__((ext_vector_type(16))) int          v16i;
typedef __attribute__((ext_vector_type(8)))  float        v8f;
typedef __attribute__((ext_vector_type(4)))  unsigned int u32x4;
typedef __attribute__((ext_vector_type(4)))  float        f32x4;
typedef __attribute__((ext_vector_type(2)))  float        f32x2;

union Frag8 {
  v16i v;
  int  u[16];
};

__device__ __forceinline__ unsigned char f32_to_e4m3(float f) {
  unsigned int u = __float_as_uint(f);
  unsigned char s = (unsigned char)((u >> 24) & 0x80u);
  unsigned int a = u & 0x7FFFFFFFu;
  if (a >= 0x43E00000u) return (unsigned char)(s | 0x7Eu);
  if (a < 0x3C800000u) {
    float av = __uint_as_float(a);
    int m = (int)(av * 512.0f + 0.5f);
    return (unsigned char)(s | (m >= 8 ? 0x08 : m));
  }
  unsigned int e = (a >> 23) - 127u + 7u;
  unsigned int man = a & 0x7FFFFFu;
  unsigned int m3 = man >> 20;
  unsigned int rest = man & 0xFFFFFu;
  m3 += (rest > 0x80000u) || (rest == 0x80000u && (m3 & 1u));
  if (m3 == 8u) { m3 = 0u; e += 1u; if (e > 15u) return (unsigned char)(s | 0x7Eu); }
  return (unsigned char)(s | (e << 3) | m3);
}

__device__ __forceinline__ float fast_sigmoid(float x) {
  return 1.f / (1.f + __expf(-x));
}
__device__ __forceinline__ float fast_tanh(float x) {
  float e = __expf(2.f * x);
  return (e - 1.f) / (e + 1.f);
}

__global__ void swizzle_whh_fp8(const float* __restrict__ whh,
                                unsigned char* __restrict__ dst) {
  int idx = blockIdx.x * blockDim.x + threadIdx.x;
  int b  = idx & 3;
  int v  = (idx >> 2) & 15;
  int L  = (idx >> 6) & 31;
  int kt = (idx >> 11) & 1;
  int nt = idx >> 12;
  int gate = nt & 3, hdg = nt >> 2;
  int n = gate * HID + hdg * 16 + (L & 15);
  int K = kt * 128 + ((v >> 2) << 5) + ((L >> 4) << 4) + ((v & 3) << 2) + b;
  dst[idx] = f32_to_e4m3(whh[n * HID + K]);
}

__global__ void features_kernel(const float* __restrict__ x,
                                const float* __restrict__ conv_w,
                                const float* __restrict__ conv_b,
                                float* __restrict__ conv_feat,
                                float* __restrict__ ksim) {
  __shared__ float rs[256];
  __shared__ float rq[256];
  int b = blockIdx.x, tid = threadIdx.x;
  float cw0 = conv_w[0], cw1 = conv_w[1], cw2 = conv_w[2], cw3 = conv_w[3];
  float cb = conv_b[0];
  float acc_s = 0.f, acc_q = 0.f;
  for (int s = tid; s < SEQ; s += 256) {
    f32x4 xv = *(const f32x4*)(x + ((size_t)b * SEQ + s) * 4);
    float d = xv[0]*cw0 + xv[1]*cw1 + xv[2]*cw2 + xv[3]*cw3 + cb;
    acc_s += fast_sigmoid(d);
    acc_q += xv[0]*xv[0] + xv[1]*xv[1] + xv[2]*xv[2] + xv[3]*xv[3];
  }
  rs[tid] = acc_s; rq[tid] = acc_q;
  __syncthreads();
  for (int st = 128; st > 0; st >>= 1) {
    if (tid < st) { rs[tid] += rs[tid + st]; rq[tid] += rq[tid + st]; }
    __syncthreads();
  }
  if (tid == 0) {
    conv_feat[b] = rs[0] * (1.f / SEQ);
    ksim[b]      = __expf(-rq[0]);
  }
}

__global__ __launch_bounds__(256)
__attribute__((amdgpu_waves_per_eu(2, 2)))
void lstm_kernel(const float* __restrict__ x,
                 const float* __restrict__ wih,
                 const float* __restrict__ bih,
                 const float* __restrict__ bhh,
                 const unsigned char* __restrict__ whh_fp8,
                 float* __restrict__ last_h) {
  __shared__ __align__(16) unsigned char hsw[2 * 4096];
  __shared__ f32x4 xs[2][16];

  const int tid  = threadIdx.x;
  const int lane = tid & 31;
  const int wv   = tid >> 5;
  const int wg   = blockIdx.x;
  const int lane15  = lane & 15;
  const int lanehi  = lane >> 4;
  const int rowbase = lanehi << 3;

  Frag8 bres[16];
#pragma unroll
  for (int f = 0; f < 16; ++f) {
    const u32x4* bp =
        (const u32x4*)whh_fp8 + (((size_t)(wv * 16 + f) * 32 + lane) * 4);
#pragma unroll
    for (int k = 0; k < 4; ++k) ((u32x4*)bres[f].u)[k] = bp[k];
  }

  f32x2 bx[8];
#pragma unroll
  for (int i = 0; i < 8; ++i) {
    const int nt = wv * 8 + i;
    const int gate = nt & 3, hdg = nt >> 2;
    const int n_w = gate * HID + hdg * 16 + lane15;
    bx[i] = *(const f32x2*)(wih + (size_t)n_w * 4 + (lanehi << 1));
  }

  float br[8];
  int   hbase[2];
  const int hd0 = (wv * 2) * 16 + lane15;
#pragma unroll
  for (int g2 = 0; g2 < 2; ++g2) {
    const int hd = hd0 + g2 * 16;
#pragma unroll
    for (int gt = 0; gt < 4; ++gt) {
      const int rw = gt * HID + hd;
      br[g2 * 4 + gt] = bih[rw] + bhh[rw];
    }
    const int kt   = hd >> 7;
    const int krem = hd & 127;
    const int sub  = krem >> 6;
    const int k64  = krem & 63;
    const int r    = ((k64 >> 5) << 2) + ((k64 >> 2) & 3);
    const int Lhi  = (k64 >> 4) & 1;
    const int bb   = k64 & 3;
    hbase[g2] = kt * 2048 + Lhi * 1024 + (sub * 8 + r) * 4 + bb;
  }

  float c[16];
#pragma unroll
  for (int i = 0; i < 16; ++i) c[i] = 0.f;

  {
    unsigned int* h32 = (unsigned int*)hsw;
#pragma unroll
    for (int i = 0; i < 8; ++i) h32[tid + i * 256] = 0u;
  }
  if (tid < 16)
    xs[0][tid] = *(const f32x4*)(x + (((size_t)(wg * 16 + tid)) * SEQ + 0) * 4);
  __syncthreads();

  v8f zero8;
#pragma unroll
  for (int e = 0; e < 8; ++e) zero8[e] = 0.f;

  for (int t = 0; t < SEQ; ++t) {
    const int p = t & 1;
    const unsigned char* hr = hsw + p * 4096;
    unsigned char*       hw = hsw + (p ^ 1) * 4096;

    if (tid < 16) {
      const int tn = (t + 1 < SEQ) ? t + 1 : t;
      xs[p ^ 1][tid] =
          *(const f32x4*)(x + (((size_t)(wg * 16 + tid)) * SEQ + tn) * 4);
    }

    f32x2 ax = *(const f32x2*)((const float*)&xs[p][lane15] + (lanehi << 1));

    Frag8 a0, a1;
    {
      const u32x4* ap0 = (const u32x4*)(hr + lane * 64);
      const u32x4* ap1 = (const u32x4*)(hr + 2048 + lane * 64);
#pragma unroll
      for (int k = 0; k < 4; ++k) ((u32x4*)a0.u)[k] = ap0[k];
#pragma unroll
      for (int k = 0; k < 4; ++k) ((u32x4*)a1.u)[k] = ap1[k];
    }

    v8f acc[8];
#pragma unroll
    for (int i = 0; i < 8; ++i) {
      acc[i] = __builtin_amdgcn_wmma_f32_16x16x4_f32(
          false, ax, false, bx[i], (short)0, zero8, false, false);
      acc[i] = __builtin_amdgcn_wmma_f32_16x16x128_fp8_fp8(
          a0.v, bres[2 * i].v, (short)0, acc[i], false, false);
      acc[i] = __builtin_amdgcn_wmma_f32_16x16x128_fp8_fp8(
          a1.v, bres[2 * i + 1].v, (short)0, acc[i], false, false);
    }

#pragma unroll
    for (int g2 = 0; g2 < 2; ++g2) {
#pragma unroll
      for (int v = 0; v < 8; ++v) {
        const int row = rowbase + v;
        float zi = acc[g2*4+0][v] + br[g2*4+0];
        float zf = acc[g2*4+1][v] + br[g2*4+1];
        float zg = acc[g2*4+2][v] + br[g2*4+2];
        float zo = acc[g2*4+3][v] + br[g2*4+3];
        float ig = fast_sigmoid(zi);
        float fg = fast_sigmoid(zf);
        float gg = fast_tanh(zg);
        float og = fast_sigmoid(zo);
        const int ci = g2 * 8 + v;
        float cn = fg * c[ci] + ig * gg;
        c[ci] = cn;
        float h = og * fast_tanh(cn);
        hw[hbase[g2] + row * 64] = f32_to_e4m3(h);
        if (t == SEQ - 1)
          last_h[((size_t)wg * 16 + row) * HID + (hd0 + g2 * 16)] = h;
      }
    }
    __syncthreads();
  }
}

__global__ void head_kernel(const float* __restrict__ last_h,
                            const float* __restrict__ conv_feat,
                            const float* __restrict__ ksim,
                            const float* __restrict__ fc_w,
                            const float* __restrict__ fc_b,
                            const float* __restrict__ out_w,
                            const float* __restrict__ out_b,
                            float* __restrict__ out) {
  __shared__ float r0[256];
  __shared__ float r1[256];
  int b = blockIdx.x, tid = threadIdx.x;
  const float* fw = fc_w + (size_t)tid * (HID + 2);
  float acc = fc_b[tid] + fw[0] * conv_feat[b] + fw[HID + 1] * ksim[b];
  const float* h = last_h + (size_t)b * HID;
  for (int j = 0; j < HID; ++j) acc += fw[1 + j] * h[j];
  float hid = acc > 0.f ? acc : 0.f;
  r0[tid] = hid * out_w[tid];
  r1[tid] = hid * out_w[HID + tid];
  __syncthreads();
  for (int st = 128; st > 0; st >>= 1) {
    if (tid < st) { r0[tid] += r0[tid + st]; r1[tid] += r1[tid + st]; }
    __syncthreads();
  }
  if (tid == 0) {
    float l0 = r0[0] + out_b[0], l1 = r1[0] + out_b[1];
    float m  = fmaxf(l0, l1);
    float e0 = __expf(l0 - m), e1 = __expf(l1 - m);
    float inv = 1.f / (e0 + e1);
    out[b * 2 + 0] = e0 * inv;
    out[b * 2 + 1] = e1 * inv;
  }
}

extern "C" void kernel_launch(void* const* d_in, const int* in_sizes, int n_in,
                              void* d_out, int out_size, void* d_ws, size_t ws_size,
                              hipStream_t stream) {
  const float* x      = (const float*)d_in[0];
  const float* conv_w = (const float*)d_in[1];
  const float* conv_b = (const float*)d_in[2];
  const float* W_ih   = (const float*)d_in[3];
  const float* W_hh   = (const float*)d_in[4];
  const float* b_ih   = (const float*)d_in[5];
  const float* b_hh   = (const float*)d_in[6];
  const float* fc_w   = (const float*)d_in[7];
  const float* fc_b   = (const float*)d_in[8];
  const float* out_w  = (const float*)d_in[9];
  const float* out_b  = (const float*)d_in[10];
  float* out = (float*)d_out;

  unsigned char* whh_fp8 = (unsigned char*)d_ws;
  float* conv_feat = (float*)((char*)d_ws + WHH_BYTES);
  float* ksim      = conv_feat + BATCH;
  float* last_h    = ksim + BATCH;

  hipLaunchKernelGGL(swizzle_whh_fp8, dim3((GATES * HID) / 256), dim3(256), 0,
                     stream, W_hh, whh_fp8);
  hipLaunchKernelGGL(features_kernel, dim3(BATCH), dim3(256), 0, stream,
                     x, conv_w, conv_b, conv_feat, ksim);
  hipLaunchKernelGGL(lstm_kernel, dim3(BATCH / 16), dim3(256), 0, stream,
                     x, W_ih, b_ih, b_hh, whh_fp8, last_h);
  hipLaunchKernelGGL(head_kernel, dim3(BATCH), dim3(256), 0, stream,
                     last_h, conv_feat, ksim, fc_w, fc_b, out_w, out_b, out);
}
